// UnconstrainedSwitchingLSTM_20907900797429
// MI455X (gfx1250) — matrix operand order repaired and checked by reading, not compiled, not run on hardware
//
#include <hip/hip_runtime.h>
#include <hip/hip_bf16.h>
#include <math.h>

typedef __bf16 v16bf __attribute__((ext_vector_type(16)));
typedef __bf16 v8bf  __attribute__((ext_vector_type(8)));
typedef float  v8f   __attribute__((ext_vector_type(8)));

#define B_SZ  32
#define T_LEN 512
#define CTRL  32
#define STATE 64
#define HID   256
#define G4H   1024  // 4*HID

// ---------- helpers ----------
__device__ __forceinline__ unsigned short f2bf_bits(float f) {
  unsigned u = __float_as_uint(f);
  u += 0x7FFFu + ((u >> 16) & 1u);           // round-to-nearest-even
  return (unsigned short)(u >> 16);
}

__device__ __forceinline__ __bf16 f2bf(float f) {
  return __builtin_bit_cast(__bf16, f2bf_bits(f));
}

__device__ __forceinline__ v8f wmma_bf16(v16bf a, v16bf b, v8f c) {
  // v_wmma_f32_16x16x32_bf16: (neg_a, A, neg_b, B, c_mod, C, reuse_a, reuse_b)
  return __builtin_amdgcn_wmma_f32_16x16x32_bf16(false, a, false, b, (short)0, c,
                                                 false, false);
}

__device__ __forceinline__ float sigm(float x) { return 1.0f / (1.0f + __expf(-x)); }

__device__ __forceinline__ v8f vzero8() {
  v8f z;
#pragma unroll
  for (int i = 0; i < 8; ++i) z[i] = 0.0f;
  return z;
}

__device__ __forceinline__ v16bf cat8(v8bf lo, v8bf hi) {
  return __builtin_shufflevector(lo, hi, 0, 1, 2, 3, 4, 5, 6, 7,
                                 8, 9, 10, 11, 12, 13, 14, 15);
}

// ---------- f32 -> bf16 weight conversion ----------
__global__ __launch_bounds__(256) void cvt_kernel(const float* __restrict__ in,
                                                  unsigned short* __restrict__ out,
                                                  int n) {
  int i = blockIdx.x * 256 + threadIdx.x;
  if (i < n) out[i] = f2bf_bits(in[i]);
}

// ---------- persistent single-workgroup LSTM layer ----------
// 512 threads = 16 waves. Wave w owns hidden columns [16w, 16w+16) for all four
// gates (i,f,g,o) and both batch tiles (M=32 -> 2 tiles of 16).
// h lives in LDS (bf16, row-major [b][k]); c lives in registers.
template <int LAYER>
__global__ __launch_bounds__(512) void lstm_kernel(
    const float* __restrict__ xin_f32,          // LAYER==0: control (B,T,CTRL)
    const unsigned short* __restrict__ xin_bf,  // LAYER==1: x0 bf16 (T,B,HID)
    const unsigned short* __restrict__ Wih,     // bf16 [4H][Kin]
    const unsigned short* __restrict__ Whh,     // bf16 [4H][HID]
    const float* __restrict__ bih, const float* __restrict__ bhh,
    unsigned short* __restrict__ xout,          // bf16 (T,B,HID)
    float* __restrict__ hout,                   // (B,HID)
    float* __restrict__ cout)                   // (B,HID)
{
  __shared__ __align__(32) unsigned short h_lds[B_SZ * HID];

  const int tid  = threadIdx.x;
  const int wv   = tid >> 5;       // wave 0..15  -> hidden tile
  const int lane = tid & 31;
  const int col  = lane & 15;      // N within tile / M within A tile
  const int half = lane >> 4;      // lane half selects K / M+8 per ISA layout
  const int nbase = wv * 16 + col; // hidden column this lane owns

  // zero-init h
  for (int i = tid; i < B_SZ * HID; i += 512) h_lds[i] = 0;
  __syncthreads();

  // biases are t-invariant
  float bias[4];
#pragma unroll
  for (int g = 0; g < 4; ++g) bias[g] = bih[g * HID + nbase] + bhh[g * HID + nbase];

  // layer0 input-weight fragments are t-invariant (K=32 -> single k-step)
  v16bf wih0[4];
  if (LAYER == 0) {
#pragma unroll
    for (int g = 0; g < 4; ++g) {
      int n = g * HID + nbase;
      wih0[g] = *(const v16bf*)&Wih[n * CTRL + half * 16];
    }
  }

  v8f creg[2];
  creg[0] = vzero8();
  creg[1] = vzero8();

  for (int t = 0; t < T_LEN; ++t) {
    v8f acc[2][4];
#pragma unroll
    for (int mt = 0; mt < 2; ++mt)
#pragma unroll
      for (int g = 0; g < 4; ++g) acc[mt][g] = vzero8();

    // ---- recurrent part: h_{t-1} @ Whh^T, K=256 ----
#pragma unroll
    for (int ko = 0; ko < HID; ko += 32) {
      v16bf afr[2];
#pragma unroll
      for (int mt = 0; mt < 2; ++mt) {
        int m = mt * 16 + col;
        v8bf lo = *(const v8bf*)&h_lds[m * HID + ko + half * 16];
        v8bf hi = *(const v8bf*)&h_lds[m * HID + ko + half * 16 + 8];
        afr[mt] = cat8(lo, hi);
      }
#pragma unroll
      for (int g = 0; g < 4; ++g) {
        int n = g * HID + nbase;
        v16bf bfr = *(const v16bf*)&Whh[n * HID + ko + half * 16];
        acc[0][g] = wmma_bf16(afr[0], bfr, acc[0][g]);
        acc[1][g] = wmma_bf16(afr[1], bfr, acc[1][g]);
      }
    }

    // ---- input part: x_t @ Wih^T ----
    if (LAYER == 0) {
      v16bf afr[2];
#pragma unroll
      for (int mt = 0; mt < 2; ++mt) {
        int b = mt * 16 + col;
        const float* p = xin_f32 + ((size_t)b * T_LEN + t) * CTRL;
        int k0 = half * 16, k1 = half * 16 + 8;
        float4 f0 = *(const float4*)(p + k0);
        float4 f1 = *(const float4*)(p + k0 + 4);
        float4 f2 = *(const float4*)(p + k1);
        float4 f3 = *(const float4*)(p + k1 + 4);
        v16bf a;
        a[0]  = f2bf(f0.x); a[1]  = f2bf(f0.y); a[2]  = f2bf(f0.z); a[3]  = f2bf(f0.w);
        a[4]  = f2bf(f1.x); a[5]  = f2bf(f1.y); a[6]  = f2bf(f1.z); a[7]  = f2bf(f1.w);
        a[8]  = f2bf(f2.x); a[9]  = f2bf(f2.y); a[10] = f2bf(f2.z); a[11] = f2bf(f2.w);
        a[12] = f2bf(f3.x); a[13] = f2bf(f3.y); a[14] = f2bf(f3.z); a[15] = f2bf(f3.w);
        afr[mt] = a;
      }
#pragma unroll
      for (int g = 0; g < 4; ++g) {
        acc[0][g] = wmma_bf16(afr[0], wih0[g], acc[0][g]);
        acc[1][g] = wmma_bf16(afr[1], wih0[g], acc[1][g]);
      }
    } else {
      const unsigned short* xt = xin_bf + (size_t)t * B_SZ * HID;
#pragma unroll
      for (int ko = 0; ko < HID; ko += 32) {
        v16bf afr[2];
#pragma unroll
        for (int mt = 0; mt < 2; ++mt) {
          int m = mt * 16 + col;
          v8bf lo = *(const v8bf*)&xt[m * HID + ko + half * 16];
          v8bf hi = *(const v8bf*)&xt[m * HID + ko + half * 16 + 8];
          afr[mt] = cat8(lo, hi);
        }
#pragma unroll
        for (int g = 0; g < 4; ++g) {
          int n = g * HID + nbase;
          v16bf bfr = *(const v16bf*)&Wih[n * HID + ko + half * 16];
          acc[0][g] = wmma_bf16(afr[0], bfr, acc[0][g]);
          acc[1][g] = wmma_bf16(afr[1], bfr, acc[1][g]);
        }
      }
    }

    // ---- gates, c/h update (all in-wave; lane holds one hidden column, 16 batches) ----
    float hf[2][8];
#pragma unroll
    for (int mt = 0; mt < 2; ++mt)
#pragma unroll
      for (int r = 0; r < 8; ++r) {
        float iv = sigm(acc[mt][0][r] + bias[0]);
        float fv = sigm(acc[mt][1][r] + bias[1]);
        float gv = tanhf(acc[mt][2][r] + bias[2]);
        float ov = sigm(acc[mt][3][r] + bias[3]);
        float cv = fv * creg[mt][r] + iv * gv;
        creg[mt][r] = cv;
        hf[mt][r] = ov * tanhf(cv);
      }

    __syncthreads();  // all waves done reading old h
#pragma unroll
    for (int mt = 0; mt < 2; ++mt)
#pragma unroll
      for (int r = 0; r < 8; ++r) {
        int b = mt * 16 + half * 8 + r;  // C/D layout: lane half -> M+8
        h_lds[b * HID + nbase] = f2bf_bits(hf[mt][r]);
      }
    __syncthreads();  // new h visible

    // coalesced copy of h_t to xout (bf16, [t][b][k])
    {
      uint4* dst = (uint4*)(xout + (size_t)t * B_SZ * HID);
      const uint4* src = (const uint4*)h_lds;
      for (int i = tid; i < (B_SZ * HID * 2) / 16; i += 512) dst[i] = src[i];
    }

    if (t == T_LEN - 1) {
#pragma unroll
      for (int mt = 0; mt < 2; ++mt)
#pragma unroll
        for (int r = 0; r < 8; ++r) {
          int b = mt * 16 + half * 8 + r;
          hout[b * HID + nbase] = hf[mt][r];
          cout[b * HID + nbase] = creg[mt][r];
        }
    }
  }
}

// ---------- big parallel GEMM: out[(b*T+t)*N + n] = x[m] @ W[n]^T + bias[n] ----------
// X: bf16 [16384][256] with m = t*32 + b.  Block computes a 64x64 f32 tile.
__global__ __launch_bounds__(256) void gen_gemm_kernel(
    const unsigned short* __restrict__ X, const unsigned short* __restrict__ W,
    const float* __restrict__ bias, float* __restrict__ out, int N) {
  __shared__ __align__(16) float tile[64 * 64];

  const int Mbase = blockIdx.x * 64;
  const int Nbase = blockIdx.y * 64;
  const int tid = threadIdx.x, wv = tid >> 5, lane = tid & 31;
  const int col = lane & 15, half = lane >> 4;
  const int mt  = wv & 3;             // 4 M-tiles
  const int nt0 = (wv >> 2) * 2;      // 2 N-tiles per wave

  v8f acc[2];
  acc[0] = vzero8();
  acc[1] = vzero8();

  const int m = Mbase + mt * 16 + col;

#pragma unroll
  for (int ko = 0; ko < HID; ko += 32) {
    v8bf lo = *(const v8bf*)&X[(size_t)m * HID + ko + half * 16];
    v8bf hi = *(const v8bf*)&X[(size_t)m * HID + ko + half * 16 + 8];
    v16bf a = cat8(lo, hi);
#pragma unroll
    for (int j = 0; j < 2; ++j) {
      int n = Nbase + (nt0 + j) * 16 + col;
      v16bf bfr = *(const v16bf*)&W[(size_t)n * HID + ko + half * 16];
      acc[j] = wmma_bf16(a, bfr, acc[j]);
    }
  }

  // stage to LDS with bias
#pragma unroll
  for (int j = 0; j < 2; ++j) {
    int nloc = (nt0 + j) * 16 + col;
    float bv = bias[Nbase + nloc];
#pragma unroll
    for (int r = 0; r < 8; ++r)
      tile[(mt * 16 + half * 8 + r) * 64 + nloc] = acc[j][r] + bv;
  }
  __syncthreads();

  // coalesced write: rows of 64 contiguous n per (t,b)
  for (int i = tid; i < 64 * 16; i += 256) {
    int row = i >> 4, c4 = i & 15;
    int mg = Mbase + row;
    int b = mg & 31, t = mg >> 5;  // m = t*32 + b
    float4 v = ((const float4*)tile)[row * 16 + c4];
    *(float4*)&out[((size_t)b * T_LEN + t) * (size_t)N + Nbase + c4 * 4] = v;
  }
}

// ---------- Bu[b][t][s] = sum_c B[b][t][s][c] * control[b][t][c] ----------
__global__ __launch_bounds__(64) void bu_kernel(const float* __restrict__ Bmat,
                                                const float* __restrict__ control,
                                                float* __restrict__ Bu) {
  __shared__ float u[CTRL];
  const int bt = blockIdx.x;  // b*T + t
  const int tid = threadIdx.x;
  if (tid < CTRL) u[tid] = control[(size_t)bt * CTRL + tid];
  __syncthreads();
  const float* row = Bmat + ((size_t)bt * STATE + tid) * CTRL;
  float s = 0.0f;
#pragma unroll
  for (int k = 0; k < CTRL; ++k) s += row[k] * u[k];
  Bu[(size_t)bt * STATE + tid] = s;
}

// ---------- sequential state recurrence: s = A_t s + Bu_t (one wave per batch) ----------
__global__ __launch_bounds__(1024) void recur_kernel(const float* __restrict__ A,
                                                     const float* __restrict__ Bu,
                                                     const float* __restrict__ s0,
                                                     float* __restrict__ states) {
  __shared__ float s_lds[B_SZ * STATE];
  const int tid = threadIdx.x;
  const int b = tid >> 5, lane = tid & 31;

  s_lds[b * STATE + lane]      = s0[b * STATE + lane];
  s_lds[b * STATE + 32 + lane] = s0[b * STATE + 32 + lane];

  for (int t = 0; t < T_LEN; ++t) {
    const size_t bt = (size_t)b * T_LEN + t;
    const float* At = A + bt * STATE * STATE;
    float y0 = Bu[bt * STATE + lane];
    float y1 = Bu[bt * STATE + 32 + lane];
    const float4* r0 = (const float4*)(At + (size_t)lane * STATE);
    const float4* r1 = (const float4*)(At + (size_t)(lane + 32) * STATE);
#pragma unroll 4
    for (int j4 = 0; j4 < STATE / 4; ++j4) {
      float4 a0 = r0[j4], a1 = r1[j4];
      float sj0 = s_lds[b * STATE + j4 * 4 + 0];
      float sj1 = s_lds[b * STATE + j4 * 4 + 1];
      float sj2 = s_lds[b * STATE + j4 * 4 + 2];
      float sj3 = s_lds[b * STATE + j4 * 4 + 3];
      y0 += a0.x * sj0 + a0.y * sj1 + a0.z * sj2 + a0.w * sj3;
      y1 += a1.x * sj0 + a1.y * sj1 + a1.z * sj2 + a1.w * sj3;
    }
    // per-wave LDS ops are in-order; other waves touch other b -> no barrier needed
    s_lds[b * STATE + lane] = y0;
    s_lds[b * STATE + 32 + lane] = y1;
    states[bt * STATE + lane] = y0;
    states[bt * STATE + 32 + lane] = y1;
  }
}

// ---------- host ----------
extern "C" void kernel_launch(void* const* d_in, const int* in_sizes, int n_in,
                              void* d_out, int out_size, void* d_ws, size_t ws_size,
                              hipStream_t stream) {
  const float* control = (const float*)d_in[0];
  const float* state0  = (const float*)d_in[1];
  const float* W_ih0 = (const float*)d_in[2];
  const float* W_hh0 = (const float*)d_in[3];
  const float* b_ih0 = (const float*)d_in[4];
  const float* b_hh0 = (const float*)d_in[5];
  const float* W_ih1 = (const float*)d_in[6];
  const float* W_hh1 = (const float*)d_in[7];
  const float* b_ih1 = (const float*)d_in[8];
  const float* b_hh1 = (const float*)d_in[9];
  const float* gAw = (const float*)d_in[10];
  const float* gAb = (const float*)d_in[11];
  const float* gBw = (const float*)d_in[12];
  const float* gBb = (const float*)d_in[13];

  // workspace layout (all offsets 64B-aligned)
  char* ws = (char*)d_ws;
  unsigned short* wih0 = (unsigned short*)(ws + 0);         //  64 KB
  unsigned short* whh0 = (unsigned short*)(ws + 65536);     // 512 KB
  unsigned short* wih1 = (unsigned short*)(ws + 589824);    // 512 KB
  unsigned short* whh1 = (unsigned short*)(ws + 1114112);   // 512 KB
  unsigned short* gawb = (unsigned short*)(ws + 1638400);   //   2 MB
  unsigned short* gbwb = (unsigned short*)(ws + 3735552);   //   1 MB
  unsigned short* x0   = (unsigned short*)(ws + 4784128);   //   8 MB bf16 (T,B,HID)
  unsigned short* x1   = (unsigned short*)(ws + 13172736);  //   8 MB bf16 (T,B,HID)
  float*          bu   = (float*)(ws + 21561344);           //   4 MB f32 (B,T,STATE)

  // d_out layout (floats): states | h | c | A | B
  float* out = (float*)d_out;
  float* out_states = out;                 // 32*512*64      = 1,048,576
  float* out_h      = out + 1048576;       // 2*32*256       =    16,384
  float* out_c      = out + 1064960;       // 2*32*256       =    16,384
  float* out_A      = out + 1081344;       // 32*512*4096    = 67,108,864
  float* out_B      = out + 68190208;      // 32*512*2048    = 33,554,432

  // 1) convert weights to bf16
  auto cvt = [&](const float* src, unsigned short* dst, int n) {
    cvt_kernel<<<(n + 255) / 256, 256, 0, stream>>>(src, dst, n);
  };
  cvt(W_ih0, wih0, G4H * CTRL);
  cvt(W_hh0, whh0, G4H * HID);
  cvt(W_ih1, wih1, G4H * HID);
  cvt(W_hh1, whh1, G4H * HID);
  cvt(gAw, gawb, STATE * STATE * HID);
  cvt(gBw, gbwb, STATE * CTRL * HID);

  // 2) LSTM layer 0 (persistent single workgroup, WMMA gates)
  lstm_kernel<0><<<1, 512, 0, stream>>>(control, nullptr, wih0, whh0, b_ih0, b_hh0,
                                        x0, out_h, out_c);
  // 3) LSTM layer 1
  lstm_kernel<1><<<1, 512, 0, stream>>>(nullptr, x0, wih1, whh1, b_ih1, b_hh1,
                                        x1, out_h + B_SZ * HID, out_c + B_SZ * HID);

  // 4) gen_A / gen_B GEMMs (bandwidth-bound on stores; WMMA math)
  gen_gemm_kernel<<<dim3(256, 64), 256, 0, stream>>>(x1, gawb, gAb, out_A,
                                                     STATE * STATE);  // N=4096
  gen_gemm_kernel<<<dim3(256, 32), 256, 0, stream>>>(x1, gbwb, gBb, out_B,
                                                     STATE * CTRL);   // N=2048

  // 5) Bu einsum
  bu_kernel<<<B_SZ * T_LEN, 64, 0, stream>>>(out_B, control, bu);

  // 6) sequential state recurrence
  recur_kernel<<<1, 1024, 0, stream>>>(out_A, bu, state0, out_states);
}
